// GNNModel_67817533604357
// MI455X (gfx1250) — compile-verified, matrix operand order checked by reading, not run on hardware
//
#include <hip/hip_runtime.h>
#include <hip/hip_bf16.h>

#define HID 128
#define NUM_REL 3
#define LN_EPS 1e-5f
#define MAT_PACK 16384          // 4 ktiles * 8 ntiles * 32 lanes * 16 elems (bf16)

typedef __attribute__((ext_vector_type(16))) __bf16 v16bf;
typedef __attribute__((ext_vector_type(8)))  float  v8f;

__device__ __forceinline__ unsigned short f2bfu(float f) {
    union { float f; unsigned u; } a; a.f = f;
    unsigned u = a.u;
    u += 0x7FFFu + ((u >> 16) & 1u);          // round-to-nearest-even
    return (unsigned short)(u >> 16);
}
__device__ __forceinline__ __bf16 us2bf(unsigned short s) {
    union { unsigned short u; __bf16 b; } c; c.u = s; return c.b;
}

__device__ __forceinline__ void atomic_add_f32(float* p, float v) {
    __hip_atomic_fetch_add(p, v, __ATOMIC_RELAXED, __HIP_MEMORY_SCOPE_AGENT);
}

// ---------------------------------------------------------------- utilities
__global__ void k_zero(float* p, long n) {
    long i = (long)blockIdx.x * blockDim.x + threadIdx.x;
    long stride = (long)gridDim.x * blockDim.x;
    for (; i < n; i += stride) p[i] = 0.0f;
}

__global__ void k_deg(const int* __restrict__ dst, float* __restrict__ rdeg, int E) {
    int e = blockIdx.x * 256 + threadIdx.x;
    if (e < E) atomic_add_f32(&rdeg[dst[e]], 1.0f);
}

__global__ void k_recip(float* rdeg, int n) {
    int i = blockIdx.x * 256 + threadIdx.x;
    if (i < n) rdeg[i] = 1.0f / fmaxf(rdeg[i], 1.0f);
}

// -------------------------------------------- weight pre-pack (bf16 B-frags)
// Matrix m = l*4 + s (s=0: lin_W[l], s=1..3: rel_W[l][s-1]).
// packed[m][kt][nt][lane][j] = bf16( W[k][col] ),
//   k = kt*32 + j + (lane>>4)*8 + (j>=8 ? 8 : 0), col = nt*16 + (lane&15)
// -> each B fragment is 16 contiguous bf16 (32 bytes) per lane.
__global__ void __launch_bounds__(256)
k_pack_w(const float* __restrict__ linW, const float* __restrict__ relW,
         unsigned short* __restrict__ pack)
{
    int m = blockIdx.x;                // 0..7
    int l = m >> 2, s = m & 3;
    const float* W = (s == 0) ? (linW + (size_t)l * HID * HID)
                              : (relW + ((size_t)l * NUM_REL + (s - 1)) * HID * HID);
    unsigned short* out = pack + (size_t)m * MAT_PACK;
    for (int i = threadIdx.x; i < MAT_PACK; i += 256) {
        int j    = i & 15;
        int lane = (i >> 4) & 31;
        int nt   = (i >> 9) & 7;
        int kt   = (i >> 12) & 3;
        int g    = lane >> 4;
        int k    = kt * 32 + j + g * 8 + ((j >= 8) ? 8 : 0);
        int col  = nt * 16 + (lane & 15);
        out[i] = f2bfu(W[(size_t)k * HID + col]);
    }
}

// ------------------------------------------- feature-build + type encoder
__global__ void __launch_bounds__(128)
k_encode(const int* __restrict__ z, const float* __restrict__ sd,
         const float* __restrict__ df, const float* __restrict__ cond,
         const float* __restrict__ mult, const int* __restrict__ ntype,
         const float* __restrict__ zemb,
         const float* __restrict__ W1, const float* __restrict__ b1,
         const float* __restrict__ W2, const float* __restrict__ b2,
         float* __restrict__ x, int N)
{
    __shared__ float raw_s[20];
    __shared__ float h1_s[HID];
    int nid = blockIdx.x;
    int j = threadIdx.x;
    if (j < 16)       raw_s[j]  = zemb[z[nid] * 16 + j];
    else if (j == 16) raw_s[16] = sd[nid];
    else if (j == 17) raw_s[17] = df[nid];
    else if (j == 18) raw_s[18] = cond[nid];
    else if (j == 19) raw_s[19] = mult[nid];
    __syncthreads();
    int t = ntype[nid];
    const float* w1 = W1 + (size_t)t * 20 * HID;
    const float* w2 = W2 + (size_t)t * HID * HID;
    float acc = b1[t * HID + j];
    #pragma unroll
    for (int k = 0; k < 20; ++k) acc = fmaf(raw_s[k], w1[k * HID + j], acc);
    h1_s[j] = fmaxf(acc, 0.0f);
    __syncthreads();
    float acc2 = b2[t * HID + j];
    for (int k = 0; k < HID; ++k) acc2 = fmaf(h1_s[k], w2[k * HID + j], acc2);
    x[(size_t)nid * HID + j] = acc2;
}

// ------------------------------------------------- per-relation edge scatter
__global__ void __launch_bounds__(256)
k_scatter(const int* __restrict__ src, const int* __restrict__ dst,
          const int* __restrict__ et, const float* __restrict__ x,
          float* __restrict__ S, int E, int N)
{
    long tid = (long)blockIdx.x * 256 + threadIdx.x;
    if (tid >= (long)E * 32) return;
    int e = (int)(tid >> 5);
    int c = (int)(tid & 31) * 4;
    int s = src[e], d = dst[e], r = et[e];
    const float4 v = *(const float4*)(x + (size_t)s * HID + c);
    float* o = S + ((size_t)r * N + d) * HID + c;
    atomic_add_f32(o + 0, v.x);
    atomic_add_f32(o + 1, v.y);
    atomic_add_f32(o + 2, v.z);
    atomic_add_f32(o + 3, v.w);
}

// --------------------------------------------------------- fused GNN layer
__global__ void __launch_bounds__(256)
k_gnn_layer(float* __restrict__ x, const float* __restrict__ S,
            const unsigned short* __restrict__ wp,   // packed B frags, 4 matrices
            const float* __restrict__ linb,
            const float* __restrict__ lng, const float* __restrict__ lnb,
            const float* __restrict__ rdeg, int N)
{
    __shared__ unsigned short a_s[4][16][HID];   // bf16 A sources
    __shared__ float h_s[16][HID];
    __shared__ float rd_s[16];
    __shared__ float mu_s[16], rs_s[16];

    const int tid = threadIdx.x;
    const int m0  = blockIdx.x * 16;

    // stage A tiles (x + 3 relation sums), f32 -> bf16
    for (int i = tid; i < 4 * 16 * HID; i += 256) {
        int s   = i >> 11;
        int rem = i & 2047;
        int row = rem >> 7;
        int k   = rem & 127;
        int node = m0 + row; if (node >= N) node = N - 1;
        float v = (s == 0) ? x[(size_t)node * HID + k]
                           : S[((size_t)(s - 1) * N + node) * HID + k];
        a_s[s][row][k] = f2bfu(v);
    }
    if (tid < 16) {
        int node = m0 + tid; if (node >= N) node = N - 1;
        rd_s[tid] = rdeg[node];
    }
    __syncthreads();

    const int lane = tid & 31;
    const int wave = tid >> 5;          // N-tile 0..7
    const int g    = lane >> 4;
    const int mn   = lane & 15;
    const int col  = wave * 16 + mn;

    v8f c_lin = {0.f,0.f,0.f,0.f,0.f,0.f,0.f,0.f};
    v8f c_agg = {0.f,0.f,0.f,0.f,0.f,0.f,0.f,0.f};

    #pragma unroll
    for (int s = 0; s < 4; ++s) {
        const unsigned short* pm = wp + (size_t)s * MAT_PACK;
        #pragma unroll
        for (int kt = 0; kt < 4; ++kt) {
            v16bf A, B;
            const int kbase = kt * 32;
            #pragma unroll
            for (int j = 0; j < 16; ++j) {
                int k = kbase + j + g * 8 + ((j >= 8) ? 8 : 0);
                A[j] = us2bf(a_s[s][mn][k]);
            }
            // one contiguous 32-byte fragment per lane
            B = *(const v16bf*)(pm + (((size_t)kt * 8 + wave) * 32 + lane) * 16);
            if (s == 0)
                c_lin = __builtin_amdgcn_wmma_f32_16x16x32_bf16(
                    false, A, false, B, (short)0, c_lin, false, false);
            else
                c_agg = __builtin_amdgcn_wmma_f32_16x16x32_bf16(
                    false, A, false, B, (short)0, c_agg, false, false);
        }
    }

    const float bc = linb[col];
    #pragma unroll
    for (int i = 0; i < 8; ++i) {
        int row = i + 8 * g;
        h_s[row][col] = c_lin[i] + bc + c_agg[i] * rd_s[row];
    }
    __syncthreads();

    if (tid < 16) {
        float sum = 0.f, sq = 0.f;
        for (int k = 0; k < HID; ++k) { float v = h_s[tid][k]; sum += v; sq += v * v; }
        float mu = sum * (1.0f / HID);
        float var = sq * (1.0f / HID) - mu * mu;
        mu_s[tid] = mu;
        rs_s[tid] = rsqrtf(var + LN_EPS);
    }
    __syncthreads();

    for (int i = tid; i < 16 * HID; i += 256) {
        int row = i >> 7, k = i & 127;
        int node = m0 + row;
        if (node < N) {
            float v = (h_s[row][k] - mu_s[row]) * rs_s[row] * lng[k] + lnb[k];
            x[(size_t)node * HID + k] = v;
        }
    }
}

// ------------------------------------------------------------ mean pool
__global__ void __launch_bounds__(128)
k_pool(const float* __restrict__ x, float* __restrict__ pooled, int N)
{
    int c = threadIdx.x;
    int n0 = blockIdx.x * 256;
    int nend = min(n0 + 256, N);
    float acc = 0.f;
    for (int n = n0; n < nend; ++n) acc += x[(size_t)n * HID + c];
    atomic_add_f32(&pooled[c], acc);
}

__global__ void k_final(const float* __restrict__ pooled,
                        const float* __restrict__ regW,
                        const float* __restrict__ regb,
                        float* __restrict__ out, int N)
{
    if (threadIdx.x == 0) {
        float inv = 1.0f / (float)N;
        float acc = 0.f;
        for (int c = 0; c < HID; ++c) acc += pooled[c] * inv * regW[c];
        out[0] = acc + regb[0];
    }
}

// ---------------------------------------------------------------- launcher
extern "C" void kernel_launch(void* const* d_in, const int* in_sizes, int n_in,
                              void* d_out, int out_size, void* d_ws, size_t ws_size,
                              hipStream_t stream)
{
    const int*   z      = (const int*)  d_in[0];
    const float* sd     = (const float*)d_in[1];
    const float* df     = (const float*)d_in[2];
    const float* cond   = (const float*)d_in[3];
    const float* mult   = (const float*)d_in[4];
    const int*   ntype  = (const int*)  d_in[5];
    const int*   eidx   = (const int*)  d_in[6];
    const int*   etype  = (const int*)  d_in[7];
    const float* zemb   = (const float*)d_in[8];
    const float* encW1  = (const float*)d_in[9];
    const float* encb1  = (const float*)d_in[10];
    const float* encW2  = (const float*)d_in[11];
    const float* encb2  = (const float*)d_in[12];
    const float* linW   = (const float*)d_in[13];
    const float* linb   = (const float*)d_in[14];
    const float* relW   = (const float*)d_in[15];
    const float* lng    = (const float*)d_in[16];
    const float* lnb    = (const float*)d_in[17];
    const float* regW   = (const float*)d_in[18];
    const float* regb   = (const float*)d_in[19];

    const int N = in_sizes[0];
    const int E = in_sizes[7];
    const int* src = eidx;
    const int* dst = eidx + E;

    // workspace: x[N*128] | S[3*N*128] | rdeg[N] | pooled[128] | wpack[8*16384 bf16]
    float* x      = (float*)d_ws;
    float* S      = x + (size_t)N * HID;
    float* rdeg   = S + (size_t)3 * N * HID;
    float* pooled = rdeg + N;
    unsigned short* wpack = (unsigned short*)(pooled + HID);

    // degree
    k_zero<<<256, 256, 0, stream>>>(rdeg, (long)N);
    k_deg<<<(E + 255) / 256, 256, 0, stream>>>(dst, rdeg, E);
    k_recip<<<(N + 255) / 256, 256, 0, stream>>>(rdeg, N);

    // pack all 8 weight matrices into WMMA B-fragment layout (bf16)
    k_pack_w<<<8, 256, 0, stream>>>(linW, relW, wpack);

    // feature build + type-specific encoder
    k_encode<<<N, 128, 0, stream>>>(z, sd, df, cond, mult, ntype, zemb,
                                    encW1, encb1, encW2, encb2, x, N);

    const long scatter_threads = (long)E * 32;
    const unsigned scatter_blocks = (unsigned)((scatter_threads + 255) / 256);
    const unsigned gemm_blocks = (unsigned)((N + 15) / 16);

    for (int l = 0; l < 2; ++l) {
        k_zero<<<4096, 256, 0, stream>>>(S, (long)3 * N * HID);
        k_scatter<<<scatter_blocks, 256, 0, stream>>>(src, dst, etype, x, S, E, N);
        k_gnn_layer<<<gemm_blocks, 256, 0, stream>>>(
            x, S,
            wpack + (size_t)l * 4 * MAT_PACK,
            linb + (size_t)l * HID,
            lng + (size_t)l * HID, lnb + (size_t)l * HID,
            rdeg, N);
    }

    k_zero<<<1, 128, 0, stream>>>(pooled, 128);
    k_pool<<<(N + 255) / 256, 128, 0, stream>>>(x, pooled, N);
    k_final<<<1, 64, 0, stream>>>(pooled, regW, regb, (float*)d_out, N);
}
